// RMVSNet_1039382085856
// MI455X (gfx1250) — matrix operand order repaired and checked by reading, not compiled, not run on hardware
//
#include <hip/hip_runtime.h>
#include <hip/hip_bf16.h>
#include <math.h>

typedef __attribute__((ext_vector_type(16))) _Float16 v16h;
typedef __attribute__((ext_vector_type(8)))  _Float16 v8h;
typedef __attribute__((ext_vector_type(8)))  float    v8f;

#define H_IMG 64
#define W_IMG 80
#define HW    5120
#define D_NUM 128
#define C_X   32

// ---------------- GRU1 (WMMA) tiling ----------------
#define TIY 8          // interior tile rows
#define TIX 16         // interior tile cols
#define TEY 12         // extended (halo 2)
#define TEX 20
#define CPAD 64        // per-pixel channel stride in LDS (32 x + 16 h + 16 pad)
#define NRPIX 180      // reset-gate region (TIY+2)*(TIX+2) = 10*18
#define NRG   12       // ceil(180/16)

__device__ __forceinline__ float sigmoidf_(float v) { return 1.f / (1.f + __expf(-v)); }

// One 48ch->16ch 3x3 conv output tile (16 pixels x 16 out-ch) as 18 WMMAs.
// A rows gathered from channels-last LDS tile; B pre-swizzled to wave32 fragment order.
__device__ __forceinline__ v8f conv48(const _Float16* sIn, const v16h* __restrict__ Bf,
                                      int py, int px, int hs, int lane)
{
  v8f acc = {};
  #pragma unroll
  for (int t = 0; t < 9; ++t) {
    const int dy = t / 3 - 1;
    const int dx = t - (t / 3) * 3 - 1;
    const _Float16* sp = sIn + ((py + dy) * TEX + (px + dx)) * CPAD;
    #pragma unroll
    for (int c32 = 0; c32 < 2; ++c32) {
      const v8h* p = (const v8h*)(sp + c32 * 32 + hs * 8);
      v8h lo = p[0];             // chunk-local K = hs*8 + 0..7
      v8h hi = p[2];             // chunk-local K = 16 + hs*8 + 0..7
      v16h a = __builtin_shufflevector(lo, hi, 0,1,2,3,4,5,6,7,8,9,10,11,12,13,14,15);
      v16h b = Bf[(t * 2 + c32) * 32 + lane];
      acc = __builtin_amdgcn_wmma_f32_16x16x32_f16(false, a, false, b, (short)0, acc,
                                                   false, false);
    }
  }
  return acc;
}

// ---------------- weight pre-swizzle into WMMA B fragment order ----------------
// B fragment: half index e = ((tap*nc32 + c32)*32 + lane)*16 + j
//   N = lane&15 (out ch), chunk-local K = (j<8 ? (lane>>4)*8 + j : 16 + (lane>>4)*8 + j-8), ch = c32*32 + K
__global__ __launch_bounds__(256) void prep_w_kernel(const float* __restrict__ w,
                                                     _Float16* __restrict__ B,
                                                     int Cout, int Cin, int nc32)
{
  int e = blockIdx.x * 256 + threadIdx.x;
  int tot = 9 * nc32 * 512;
  if (e >= tot) return;
  int j    = e & 15;
  int lane = (e >> 4) & 31;
  int kc   = e >> 9;
  int tap  = kc / nc32;
  int c32  = kc - tap * nc32;
  int N  = lane & 15;
  int hs = lane >> 4;
  int ch = c32 * 32 + (j < 8 ? hs * 8 + j : 16 + hs * 8 + (j - 8));
  float v = 0.f;
  if (N < Cout && ch < Cin) v = w[(N * Cin + ch) * 9 + tap];
  B[e] = (_Float16)v;
}

// ---------------- cost volume: x = -(E[w^2] - E[w]^2), channels-last fp16 ----------------
__global__ __launch_bounds__(256) void cost_kernel(const float* __restrict__ warped,
                                                   _Float16* __restrict__ X)
{
  size_t gid = (size_t)blockIdx.x * 256 + threadIdx.x;
  const size_t TOT = (size_t)D_NUM * C_X * HW;
  if (gid >= TOT) return;
  int x = (int)(gid % W_IMG);
  int y = (int)((gid / W_IMG) % H_IMG);
  int c = (int)((gid / HW) % C_X);
  int d = (int)(gid / ((size_t)HW * C_X));
  const size_t vs = (size_t)C_X * D_NUM * HW;          // view stride
  const size_t base = ((size_t)c * D_NUM + d) * HW + (size_t)y * W_IMG + x;
  float s1 = 0.f, s2 = 0.f;
  #pragma unroll
  for (int v = 0; v < 3; ++v) {
    float t = warped[v * vs + base];
    s1 += t; s2 += t * t;
  }
  float mean = s1 * (1.f / 3.f);
  float xv = -(s2 * (1.f / 3.f) - mean * mean);
  X[((size_t)d * HW + (size_t)y * W_IMG + x) * C_X + c] = (_Float16)xv;
}

// ---------------- zero initial hidden states ----------------
__global__ __launch_bounds__(256) void init_state_kernel(_Float16* h1, float* h2, float* h3)
{
  int i = blockIdx.x * 256 + threadIdx.x;
  if (i < HW * 16) h1[i] = (_Float16)0;
  if (i < HW * 4)  h2[i] = 0.f;
  if (i < HW * 2)  h3[i] = 0.f;
}

// ---------------- GRU1: 32 in / 16 hidden, WMMA implicit GEMM ----------------
__global__ __launch_bounds__(256) void gru1_step_kernel(
    const _Float16* __restrict__ Xd, const _Float16* __restrict__ Xn,
    const _Float16* __restrict__ H1in, _Float16* __restrict__ H1out,
    const v16h* __restrict__ Br, const v16h* __restrict__ Bu, const v16h* __restrict__ Bc,
    const float* __restrict__ br, const float* __restrict__ bu, const float* __restrict__ bc)
{
  __shared__ __align__(32) _Float16 sIn[TEY * TEX * CPAD];  // 30,720 B
  const int tid  = threadIdx.x;
  const int wave = tid >> 5;
  const int lane = tid & 31;
  const int hs   = lane >> 4;
  const int nsel = lane & 15;
  const int tx = blockIdx.x % 5, ty = blockIdx.x / 5;

  // Phase 0: cooperative channels-last tile load (x | h | zero-pad), halo 2, SAME-pad zeros
  v8h z = {};
  for (int p = tid; p < TEY * TEX; p += 256) {
    const int py = p / TEX, px = p - (p / TEX) * TEX;
    const int gy = ty * TIY + py - 2, gx = tx * TIX + px - 2;
    v8h* dv = (v8h*)(sIn + p * CPAD);
    if ((unsigned)gy < (unsigned)H_IMG && (unsigned)gx < (unsigned)W_IMG) {
      const v8h* sx = (const v8h*)(Xd + ((size_t)gy * W_IMG + gx) * C_X);
      const v8h* sh = (const v8h*)(H1in + ((size_t)gy * W_IMG + gx) * 16);
      dv[0] = sx[0]; dv[1] = sx[1]; dv[2] = sx[2]; dv[3] = sx[3];
      dv[4] = sh[0]; dv[5] = sh[1];
      dv[6] = z;     dv[7] = z;
      __builtin_prefetch(Xn + ((size_t)gy * W_IMG + gx) * C_X, 0, 3);  // next depth slice
    } else {
      dv[0]=z; dv[1]=z; dv[2]=z; dv[3]=z; dv[4]=z; dv[5]=z; dv[6]=z; dv[7]=z;
    }
  }
  __syncthreads();

  // Phase 1: reset gate r over interior+1 ring (10x18), kept in registers
  float rv[2][8] = {};
  #pragma unroll
  for (int rep = 0; rep < 2; ++rep) {
    const int g = wave + rep * 8;           // wave-uniform guard: EXEC stays full for WMMA
    if (g < NRG) {
      int idxA = g * 16 + nsel; if (idxA >= NRPIX) idxA = NRPIX - 1;
      const int py = 1 + idxA / 18, px = 1 + idxA % 18;
      v8f acc = conv48(sIn, Br, py, px, hs, lane);
      const float bb = br[nsel];
      #pragma unroll
      for (int i = 0; i < 8; ++i) rv[rep][i] = sigmoidf_(acc[i] + bb);
    }
  }

  // Phase 2: update gate u over interior (one 16-px row group per wave) + capture h_old
  float uv[8], hold[8];
  {
    const int g = wave;
    v8f acc = conv48(sIn, Bu, 2 + g, 2 + nsel, hs, lane);
    const float bb = bu[nsel];
    #pragma unroll
    for (int i = 0; i < 8; ++i) {
      uv[i] = sigmoidf_(acc[i] + bb);
      const int m = i + 8 * hs;            // D-layout row
      hold[i] = (float)sIn[((2 + g) * TEX + (2 + m)) * CPAD + 32 + nsel];
    }
  }
  __syncthreads();

  // Phase 3: overwrite h channels in LDS with r*h (only where candidate conv reads)
  #pragma unroll
  for (int rep = 0; rep < 2; ++rep) {
    const int g = wave + rep * 8;
    if (g < NRG) {
      #pragma unroll
      for (int i = 0; i < 8; ++i) {
        const int m = i + 8 * hs;
        const int idx = g * 16 + m;
        if (idx < NRPIX) {
          const int py = 1 + idx / 18, px = 1 + idx % 18;
          const int o = (py * TEX + px) * CPAD + 32 + nsel;
          sIn[o] = (_Float16)(rv[rep][i] * (float)sIn[o]);
        }
      }
    }
  }
  __syncthreads();

  // Phase 4: candidate c = tanh(conv(x, r*h)), blend, store h1'
  {
    const int g = wave;
    v8f acc = conv48(sIn, Bc, 2 + g, 2 + nsel, hs, lane);
    const float bb = bc[nsel];
    #pragma unroll
    for (int i = 0; i < 8; ++i) {
      const int m = i + 8 * hs;
      const float ct = tanhf(acc[i] + bb);
      const float hn = uv[i] * hold[i] + (1.f - uv[i]) * ct;
      const int gy = ty * TIY + g, gx = tx * TIX + m;
      H1out[((size_t)gy * W_IMG + gx) * 16 + nsel] = (_Float16)hn;
    }
  }
}

// ---------------- GRU2: 16 in / 4 hidden (scalar VALU; WMMA would be >75% zero-pad) ----------------
__global__ __launch_bounds__(256) void gru2_step_kernel(
    const _Float16* __restrict__ H1n,
    const float* __restrict__ H2in, float* __restrict__ H2out,
    const float* __restrict__ wr, const float* __restrict__ br,
    const float* __restrict__ wu, const float* __restrict__ bu,
    const float* __restrict__ wc, const float* __restrict__ bc)
{
  __shared__ _Float16 s1[20 * 20 * 16];
  __shared__ float    s2[20 * 20 * 4];
  __shared__ float    rh[20 * 20 * 4];
  __shared__ float    wL[3 * 720];
  const int tid = threadIdx.x;
  const int tx = blockIdx.x % 5, ty = blockIdx.x / 5;

  for (int i = tid; i < 720; i += 256) { wL[i] = wr[i]; wL[720 + i] = wu[i]; wL[1440 + i] = wc[i]; }
  for (int p = tid; p < 400; p += 256) {
    const int py = p / 20, px = p - (p / 20) * 20;
    const int gy = ty * 16 + py - 2, gx = tx * 16 + px - 2;
    if ((unsigned)gy < (unsigned)H_IMG && (unsigned)gx < (unsigned)W_IMG) {
      const _Float16* s = H1n + ((size_t)gy * W_IMG + gx) * 16;
      #pragma unroll
      for (int c = 0; c < 16; ++c) s1[p * 16 + c] = s[c];
      const float* s2g = H2in + ((size_t)gy * W_IMG + gx) * 4;
      #pragma unroll
      for (int c = 0; c < 4; ++c) s2[p * 4 + c] = s2g[c];
    } else {
      #pragma unroll
      for (int c = 0; c < 16; ++c) s1[p * 16 + c] = (_Float16)0;
      #pragma unroll
      for (int c = 0; c < 4; ++c) s2[p * 4 + c] = 0.f;
    }
  }
  __syncthreads();

  // reset gate on interior+1 ring (18x18), store r*h2
  for (int idx = tid; idx < 324; idx += 256) {
    const int py = 1 + idx / 18, px = 1 + idx % 18;
    float acc[4] = { br[0], br[1], br[2], br[3] };
    #pragma unroll
    for (int t = 0; t < 9; ++t) {
      const int q = (py + t / 3 - 1) * 20 + (px + t % 3 - 1);
      for (int ic = 0; ic < 16; ++ic) {
        const float v = (float)s1[q * 16 + ic];
        #pragma unroll
        for (int oc = 0; oc < 4; ++oc) acc[oc] += v * wL[(oc * 20 + ic) * 9 + t];
      }
      #pragma unroll
      for (int ic = 0; ic < 4; ++ic) {
        const float v = s2[q * 4 + ic];
        #pragma unroll
        for (int oc = 0; oc < 4; ++oc) acc[oc] += v * wL[(oc * 20 + 16 + ic) * 9 + t];
      }
    }
    const int pp = py * 20 + px;
    #pragma unroll
    for (int oc = 0; oc < 4; ++oc) rh[pp * 4 + oc] = sigmoidf_(acc[oc]) * s2[pp * 4 + oc];
  }
  __syncthreads();

  // u, c, blend on interior (one pixel per thread)
  {
    const int py = 2 + (tid >> 4), px = 2 + (tid & 15);
    float ua[4] = { bu[0], bu[1], bu[2], bu[3] };
    float ca[4] = { bc[0], bc[1], bc[2], bc[3] };
    #pragma unroll
    for (int t = 0; t < 9; ++t) {
      const int q = (py + t / 3 - 1) * 20 + (px + t % 3 - 1);
      for (int ic = 0; ic < 16; ++ic) {
        const float v = (float)s1[q * 16 + ic];
        #pragma unroll
        for (int oc = 0; oc < 4; ++oc) {
          ua[oc] += v * wL[720 + (oc * 20 + ic) * 9 + t];
          ca[oc] += v * wL[1440 + (oc * 20 + ic) * 9 + t];
        }
      }
      #pragma unroll
      for (int ic = 0; ic < 4; ++ic) {
        const float vh = s2[q * 4 + ic];
        const float vr = rh[q * 4 + ic];
        #pragma unroll
        for (int oc = 0; oc < 4; ++oc) {
          ua[oc] += vh * wL[720 + (oc * 20 + 16 + ic) * 9 + t];
          ca[oc] += vr * wL[1440 + (oc * 20 + 16 + ic) * 9 + t];
        }
      }
    }
    const int gy = ty * 16 + py - 2, gx = tx * 16 + px - 2;
    const int pp = py * 20 + px;
    #pragma unroll
    for (int oc = 0; oc < 4; ++oc) {
      const float u = sigmoidf_(ua[oc]);
      const float ct = tanhf(ca[oc]);
      H2out[((size_t)gy * W_IMG + gx) * 4 + oc] = u * s2[pp * 4 + oc] + (1.f - u) * ct;
    }
  }
}

// ---------------- GRU3 (4 in / 2 hidden) fused with projection conv -> regs[d] ----------------
__global__ __launch_bounds__(256) void gru3_step_kernel(
    const float* __restrict__ H2n,
    const float* __restrict__ H3in, float* __restrict__ H3out,
    const float* __restrict__ wr, const float* __restrict__ br,
    const float* __restrict__ wu, const float* __restrict__ bu,
    const float* __restrict__ wc, const float* __restrict__ bc,
    const float* __restrict__ wp, const float* __restrict__ bp,
    float* __restrict__ regd)
{
  __shared__ float s2[22 * 22 * 4];
  __shared__ float s3[22 * 22 * 2];
  __shared__ float rh[22 * 22 * 2];
  __shared__ float sn[22 * 22 * 2];
  __shared__ float wL[3 * 108 + 18];
  const int tid = threadIdx.x;
  const int tx = blockIdx.x % 5, ty = blockIdx.x / 5;

  for (int i = tid; i < 108; i += 256) { wL[i] = wr[i]; wL[108 + i] = wu[i]; wL[216 + i] = wc[i]; }
  if (tid < 18) wL[324 + tid] = wp[tid];
  for (int p = tid; p < 484; p += 256) {
    const int py = p / 22, px = p - (p / 22) * 22;
    const int gy = ty * 16 + py - 3, gx = tx * 16 + px - 3;
    const bool inb = (unsigned)gy < (unsigned)H_IMG && (unsigned)gx < (unsigned)W_IMG;
    #pragma unroll
    for (int c = 0; c < 4; ++c) s2[p * 4 + c] = inb ? H2n[((size_t)gy * W_IMG + gx) * 4 + c] : 0.f;
    #pragma unroll
    for (int c = 0; c < 2; ++c) s3[p * 2 + c] = inb ? H3in[((size_t)gy * W_IMG + gx) * 2 + c] : 0.f;
  }
  __syncthreads();

  // reset gate on interior+2 ring (20x20)
  for (int idx = tid; idx < 400; idx += 256) {
    const int py = 1 + idx / 20, px = 1 + idx % 20;
    float acc[2] = { br[0], br[1] };
    #pragma unroll
    for (int t = 0; t < 9; ++t) {
      const int q = (py + t / 3 - 1) * 22 + (px + t % 3 - 1);
      #pragma unroll
      for (int ic = 0; ic < 4; ++ic) {
        const float v = s2[q * 4 + ic];
        acc[0] += v * wL[(0 * 6 + ic) * 9 + t];
        acc[1] += v * wL[(1 * 6 + ic) * 9 + t];
      }
      #pragma unroll
      for (int ic = 0; ic < 2; ++ic) {
        const float v = s3[q * 2 + ic];
        acc[0] += v * wL[(0 * 6 + 4 + ic) * 9 + t];
        acc[1] += v * wL[(1 * 6 + 4 + ic) * 9 + t];
      }
    }
    const int pp = py * 22 + px;
    #pragma unroll
    for (int oc = 0; oc < 2; ++oc) rh[pp * 2 + oc] = sigmoidf_(acc[oc]) * s3[pp * 2 + oc];
  }
  __syncthreads();

  // u, c, h3' on interior+1 ring (18x18); store h3' to LDS (+global on interior)
  for (int idx = tid; idx < 324; idx += 256) {
    const int py = 2 + idx / 18, px = 2 + idx % 18;
    float ua[2] = { bu[0], bu[1] };
    float ca[2] = { bc[0], bc[1] };
    #pragma unroll
    for (int t = 0; t < 9; ++t) {
      const int q = (py + t / 3 - 1) * 22 + (px + t % 3 - 1);
      #pragma unroll
      for (int ic = 0; ic < 4; ++ic) {
        const float v = s2[q * 4 + ic];
        ua[0] += v * wL[108 + (0 * 6 + ic) * 9 + t];
        ua[1] += v * wL[108 + (1 * 6 + ic) * 9 + t];
        ca[0] += v * wL[216 + (0 * 6 + ic) * 9 + t];
        ca[1] += v * wL[216 + (1 * 6 + ic) * 9 + t];
      }
      #pragma unroll
      for (int ic = 0; ic < 2; ++ic) {
        const float vh = s3[q * 2 + ic];
        const float vr = rh[q * 2 + ic];
        ua[0] += vh * wL[108 + (0 * 6 + 4 + ic) * 9 + t];
        ua[1] += vh * wL[108 + (1 * 6 + 4 + ic) * 9 + t];
        ca[0] += vr * wL[216 + (0 * 6 + 4 + ic) * 9 + t];
        ca[1] += vr * wL[216 + (1 * 6 + 4 + ic) * 9 + t];
      }
    }
    const int pp = py * 22 + px;
    const bool interior = (py >= 3) && (py <= 18) && (px >= 3) && (px <= 18);
    const int gy = ty * 16 + py - 3, gx = tx * 16 + px - 3;
    #pragma unroll
    for (int oc = 0; oc < 2; ++oc) {
      const float u = sigmoidf_(ua[oc]);
      const float ct = tanhf(ca[oc]);
      const float hn = u * s3[pp * 2 + oc] + (1.f - u) * ct;
      sn[pp * 2 + oc] = hn;
      if (interior) H3out[((size_t)gy * W_IMG + gx) * 2 + oc] = hn;
    }
  }
  __syncthreads();

  // projection conv -> reg map for this depth
  {
    const int py = 3 + (tid >> 4), px = 3 + (tid & 15);
    float acc = bp[0];
    #pragma unroll
    for (int t = 0; t < 9; ++t) {
      const int q = (py + t / 3 - 1) * 22 + (px + t % 3 - 1);
      acc += sn[q * 2 + 0] * wL[324 + 0 * 9 + t];
      acc += sn[q * 2 + 1] * wL[324 + 1 * 9 + t];
    }
    const int gy = ty * 16 + py - 3, gx = tx * 16 + px - 3;
    regd[(size_t)gy * W_IMG + gx] = acc;
  }
}

// ---------------- softmax over depth per pixel ----------------
__global__ __launch_bounds__(256) void softmax_kernel(const float* __restrict__ regs,
                                                      float* __restrict__ out)
{
  const int p = blockIdx.x * 256 + threadIdx.x;
  if (p >= HW) return;
  float m = -1e30f, s = 0.f;
  for (int d = 0; d < D_NUM; ++d) {
    const float v = regs[(size_t)d * HW + p];
    if (v > m) { s = s * __expf(m - v) + 1.f; m = v; }
    else       { s += __expf(v - m); }
  }
  const float inv = 1.f / s;
  for (int d = 0; d < D_NUM; ++d) {
    const float v = regs[(size_t)d * HW + p];
    out[(size_t)d * HW + p] = __expf(v - m) * inv;
  }
}

// ---------------- host orchestration (graph-capture safe: only launches) ----------------
extern "C" void kernel_launch(void* const* d_in, const int* in_sizes, int n_in,
                              void* d_out, int out_size, void* d_ws, size_t ws_size,
                              hipStream_t stream)
{
  (void)in_sizes; (void)n_in; (void)out_size;
  const float* warped = (const float*)d_in[0];
  const float* wr1 = (const float*)d_in[1];  const float* br1 = (const float*)d_in[2];
  const float* wu1 = (const float*)d_in[3];  const float* bu1 = (const float*)d_in[4];
  const float* wc1 = (const float*)d_in[5];  const float* bc1 = (const float*)d_in[6];
  const float* wr2 = (const float*)d_in[7];  const float* br2 = (const float*)d_in[8];
  const float* wu2 = (const float*)d_in[9];  const float* bu2 = (const float*)d_in[10];
  const float* wc2 = (const float*)d_in[11]; const float* bc2 = (const float*)d_in[12];
  const float* wr3 = (const float*)d_in[13]; const float* br3 = (const float*)d_in[14];
  const float* wu3 = (const float*)d_in[15]; const float* bu3 = (const float*)d_in[16];
  const float* wc3 = (const float*)d_in[17]; const float* bc3 = (const float*)d_in[18];
  const float* wp  = (const float*)d_in[19]; const float* bp  = (const float*)d_in[20];
  float* out = (float*)d_out;

  char* ws = (char*)d_ws;
  size_t off = 0;
  _Float16* X    = (_Float16*)(ws + off); off += (size_t)D_NUM * HW * C_X * sizeof(_Float16);
  float*    regs = (float*)(ws + off);    off += (size_t)D_NUM * HW * sizeof(float);
  _Float16* H1   = (_Float16*)(ws + off); off += (size_t)2 * HW * 16 * sizeof(_Float16);
  float*    H2   = (float*)(ws + off);    off += (size_t)2 * HW * 4 * sizeof(float);
  float*    H3   = (float*)(ws + off);    off += (size_t)2 * HW * 2 * sizeof(float);
  _Float16* Br   = (_Float16*)(ws + off); off += (size_t)9216 * sizeof(_Float16);
  _Float16* Bu   = (_Float16*)(ws + off); off += (size_t)9216 * sizeof(_Float16);
  _Float16* Bc   = (_Float16*)(ws + off); off += (size_t)9216 * sizeof(_Float16);
  if (ws_size < off) return;  // ~43.1 MiB required

  // one-time: swizzle GRU1 weights into WMMA B-fragment order
  prep_w_kernel<<<36, 256, 0, stream>>>(wr1, Br, 16, 48, 2);
  prep_w_kernel<<<36, 256, 0, stream>>>(wu1, Bu, 16, 48, 2);
  prep_w_kernel<<<36, 256, 0, stream>>>(wc1, Bc, 16, 48, 2);

  // variance cost volume -> channels-last fp16 (memory-bound: 252 MB read)
  cost_kernel<<<(int)(((size_t)D_NUM * HW * C_X + 255) / 256), 256, 0, stream>>>(warped, X);
  init_state_kernel<<<(HW * 16 + 255) / 256, 256, 0, stream>>>(H1, H2, H3);

  for (int d = 0; d < D_NUM; ++d) {
    const _Float16* Xd = X + (size_t)d * HW * C_X;
    const _Float16* Xn = X + (size_t)((d + 1 < D_NUM) ? d + 1 : d) * HW * C_X;
    gru1_step_kernel<<<40, 256, 0, stream>>>(
        Xd, Xn,
        H1 + (size_t)(d & 1) * HW * 16, H1 + (size_t)((d + 1) & 1) * HW * 16,
        (const v16h*)Br, (const v16h*)Bu, (const v16h*)Bc, br1, bu1, bc1);
    gru2_step_kernel<<<20, 256, 0, stream>>>(
        H1 + (size_t)((d + 1) & 1) * HW * 16,
        H2 + (size_t)(d & 1) * HW * 4, H2 + (size_t)((d + 1) & 1) * HW * 4,
        wr2, br2, wu2, bu2, wc2, bc2);
    gru3_step_kernel<<<20, 256, 0, stream>>>(
        H2 + (size_t)((d + 1) & 1) * HW * 4,
        H3 + (size_t)(d & 1) * HW * 2, H3 + (size_t)((d + 1) & 1) * HW * 2,
        wr3, br3, wu3, bu3, wc3, bc3, wp, bp, regs + (size_t)d * HW);
  }

  softmax_kernel<<<(HW + 255) / 256, 256, 0, stream>>>(regs, out);
}
